// AutoEncoder_55044300865887
// MI455X (gfx1250) — matrix operand order repaired and checked by reading, not compiled, not run on hardware
//
#include <hip/hip_runtime.h>
#include <hip/hip_bf16.h>

#define DIM    128
#define HID    256
#define NBATCH 2048
#define NTOT   32768
#define NMAX   32
#define VMID   192   // encoder val_net hidden
#define SMID   128   // size head hidden
#define DMID   192   // decoder hidden
#define SEG_CAP 128
#define ENC_M  32    // rows per encoder workgroup
#define DEC_M  64    // rows per decoder workgroup (2 sets)

typedef __bf16 bf16_t;
typedef bf16_t v16bf __attribute__((ext_vector_type(16)));
typedef bf16_t v8bf  __attribute__((ext_vector_type(8)));
typedef float  v8f   __attribute__((ext_vector_type(8)));

__device__ __forceinline__ float mishf(float v) {
  float sp = (v > 20.f) ? v : __logf(1.f + __expf(v));
  return v * tanhf(sp);
}

// ---- WMMA fragment builders (CDNA5 16x16x32 bf16 layouts, wave32) ----
// A 16x32 (MxK): lane half h, row m. v16bf element order is exactly
// K = [k0+8h .. k0+8h+7] then [k0+8h+16 .. k0+8h+23]  => two 16B LDS loads.
__device__ __forceinline__ v16bf load_A(const bf16_t* base, int ldk, int k0, int lane) {
  const int half = lane >> 4, m = lane & 15;
  const bf16_t* p = base + m * ldk + k0 + half * 8;
  v8bf lo = *(const v8bf*)(p);
  v8bf hi = *(const v8bf*)(p + 16);
  return __builtin_shufflevector(lo, hi, 0, 1, 2, 3, 4, 5, 6, 7, 8, 9, 10, 11, 12, 13, 14, 15);
}
// B 32x16 (KxN): pre-packed fragment-major. P[((tile*KB+kb)*32+lane)*16 + e]
// holds W[kb*32 + (lane>>4)*8 + (e&7) + (e>>3)*16][tile*16 + (lane&15)].  => two 16B global loads.
__device__ __forceinline__ v16bf load_Bp(const bf16_t* __restrict__ P, int KB, int kb, int tile,
                                         int lane) {
  const bf16_t* p = P + (((size_t)(tile * KB + kb) * 32 + lane) << 4);
  v8bf lo = *(const v8bf*)(p);
  v8bf hi = *(const v8bf*)(p + 8);
  return __builtin_shufflevector(lo, hi, 0, 1, 2, 3, 4, 5, 6, 7, 8, 9, 10, 11, 12, 13, 14, 15);
}

#define WMMA_BF16(acc, a, b) \
  __builtin_amdgcn_wmma_f32_16x16x32_bf16(false, (a), false, (b), (short)0, (acc), false, false)

// -------- prep: pack f32 weights [KxN] into bf16 WMMA-B fragment order --------
__global__ void k_pack_w(const float* __restrict__ W, bf16_t* __restrict__ P, int K, int N) {
  int i = blockIdx.x * blockDim.x + threadIdx.x;
  if (i >= K * N) return;
  const int KB = K >> 5;
  const int e = i & 15;
  const int l = (i >> 4) & 31;
  const int rest = i >> 9;
  const int kb = rest % KB;
  const int t = rest / KB;
  const int k = kb * 32 + (l >> 4) * 8 + (e & 7) + (e >> 3) * 16;
  const int n = t * 16 + (l & 15);
  P[i] = (bf16_t)W[k * N + n];
}

// ---------------- rank: mag = x @ rank_w + rank_b ----------------
__global__ void k_rank(const float* __restrict__ x, const float* __restrict__ rank_w,
                       const float* __restrict__ rank_b, float* __restrict__ mag) {
  const int lane = threadIdx.x & 31;
  const int wave = threadIdx.x >> 5;
  const int row = blockIdx.x * 8 + wave;
  float s = 0.f;
  for (int d = lane; d < DIM; d += 32) s += x[row * DIM + d] * rank_w[d];
#pragma unroll
  for (int off = 16; off > 0; off >>= 1) s += __shfl_down(s, off, 32);
  if (lane == 0) mag[row] = s + rank_b[0];
}

// ---------------- segment bounds (batch is sorted int32) ----------------
__global__ void k_segbounds(const int* __restrict__ batch, int* __restrict__ seg_start,
                            int* __restrict__ ncnt) {
  int b = blockIdx.x * blockDim.x + threadIdx.x;
  if (b >= NBATCH) return;
  int lo = 0, hi = NTOT;
  while (lo < hi) { int m = (lo + hi) >> 1; if (batch[m] < b) lo = m + 1; else hi = m; }
  int s0 = lo;
  lo = s0; hi = NTOT;
  while (lo < hi) { int m = (lo + hi) >> 1; if (batch[m] < b + 1) lo = m + 1; else hi = m; }
  seg_start[b] = s0;
  ncnt[b] = lo - s0;
}

// ---------------- z init: z = n*card_w + card_b (real), z_im = 0 ----------------
__global__ void k_zinit(const int* __restrict__ ncnt, const float* __restrict__ card_w,
                        const float* __restrict__ card_b, float* __restrict__ z_re,
                        float* __restrict__ z_im) {
  int i = blockIdx.x * blockDim.x + threadIdx.x;
  if (i >= NBATCH * HID) return;
  int b = i >> 8, j = i & (HID - 1);
  z_re[i] = (float)ncnt[b] * card_w[j] + card_b[j];
  z_im[i] = 0.f;
}

// ---------------- stable in-segment sort by mag (O(L^2) rank) ----------------
__global__ void k_sort(const float* __restrict__ mag, const int* __restrict__ seg_start,
                       const int* __restrict__ ncnt, int* __restrict__ order) {
  __shared__ float sm[SEG_CAP];
  const int b = blockIdx.x;
  const int s0 = seg_start[b];
  const int L = ncnt[b];
  if (L > SEG_CAP) {  // block-uniform fallback (never expected: E[L]=16)
    for (int j = threadIdx.x; j < L; j += blockDim.x) order[s0 + j] = s0 + j;
    return;
  }
  for (int i = threadIdx.x; i < L; i += blockDim.x) sm[i] = mag[s0 + i];
  __syncthreads();
  for (int j = threadIdx.x; j < L; j += blockDim.x) {
    const float mj = sm[j];
    int r = 0;
    for (int i = 0; i < L; ++i) {
      const float mi = sm[i];
      r += (mi < mj) ? 1 : ((mi == mj && i < j) ? 1 : 0);
    }
    order[s0 + r] = s0 + j;
  }
}

// ---------------- encoder: ENC_M sorted rows per workgroup ----------------
__global__ __launch_bounds__(256) void k_encoder(
    const float* __restrict__ x, const int* __restrict__ batch,
    const int* __restrict__ order, const int* __restrict__ seg_start,
    const bf16_t* __restrict__ wv1p, const float* __restrict__ b1,
    const float* __restrict__ ln_g, const float* __restrict__ ln_b,
    const bf16_t* __restrict__ wv2p, const float* __restrict__ b2,
    float* __restrict__ z_re, float* __restrict__ z_im) {
  __shared__ __align__(16) bf16_t sA[ENC_M][DIM];     // 8KB
  __shared__ float sO1[ENC_M][VMID];                  // 24KB
  __shared__ __align__(16) bf16_t sH1[ENC_M][VMID];   // 12KB
  __shared__ float sH[ENC_M][HID];                    // 32KB
  __shared__ float sPart[ENC_M][8];
  __shared__ float sMu[ENC_M], sVr[ENC_M];

  const int tid = threadIdx.x;
  const int lane = tid & 31, wave = tid >> 5;
  const int R0 = blockIdx.x * ENC_M;

  for (int idx = tid; idx < ENC_M * DIM; idx += 256) {
    const int r = idx >> 7, d = idx & (DIM - 1);
    const int src = order[R0 + r];
    sA[r][d] = (bf16_t)x[src * DIM + d];
  }
  __syncthreads();

  // GEMM1: [ENC_M x 128] @ [128 x 192]; B fragments held in regs across M-tiles
  for (int tile = wave; tile < VMID / 16; tile += 8) {
    v16bf bf[DIM / 32];
#pragma unroll
    for (int kb = 0; kb < DIM / 32; ++kb) bf[kb] = load_Bp(wv1p, DIM / 32, kb, tile, lane);
    const int half = lane >> 4, nn = tile * 16 + (lane & 15);
    const float bias = b1[nn];
#pragma unroll
    for (int mt = 0; mt < ENC_M / 16; ++mt) {
      v8f acc = {};
#pragma unroll
      for (int kb = 0; kb < DIM / 32; ++kb)
        acc = WMMA_BF16(acc, load_A(&sA[mt * 16][0], DIM, kb * 32, lane), bf[kb]);
#pragma unroll
      for (int r = 0; r < 8; ++r) sO1[mt * 16 + r + half * 8][nn] = acc[r] + bias;
    }
  }
  __syncthreads();

  // LayerNorm stats: 8 threads per row, LDS partial reduce
  {
    const int r = tid >> 3, c = tid & 7;
    float p = 0.f;
    for (int t = c; t < VMID; t += 8) p += sO1[r][t];
    sPart[r][c] = p;
  }
  __syncthreads();
  if (tid < ENC_M) {
    float mu = 0.f;
#pragma unroll
    for (int i = 0; i < 8; ++i) mu += sPart[tid][i];
    sMu[tid] = mu * (1.f / VMID);
  }
  __syncthreads();
  {
    const int r = tid >> 3, c = tid & 7;
    const float mu = sMu[r];
    float p = 0.f;
    for (int t = c; t < VMID; t += 8) { float d = sO1[r][t] - mu; p += d * d; }
    sPart[r][c] = p;
  }
  __syncthreads();
  if (tid < ENC_M) {
    float v = 0.f;
#pragma unroll
    for (int i = 0; i < 8; ++i) v += sPart[tid][i];
    sVr[tid] = v * (1.f / VMID);
  }
  __syncthreads();
  for (int idx = tid; idx < ENC_M * VMID; idx += 256) {
    const int r = idx / VMID, j = idx % VMID;
    const float xv = (sO1[r][j] - sMu[r]) * rsqrtf(sVr[r] + 1e-5f) * ln_g[j] + ln_b[j];
    sH1[r][j] = (bf16_t)mishf(xv);
  }
  __syncthreads();

  // GEMM2: [ENC_M x 192] @ [192 x 256]
  for (int tile = wave; tile < HID / 16; tile += 8) {
    v16bf bf[VMID / 32];
#pragma unroll
    for (int kb = 0; kb < VMID / 32; ++kb) bf[kb] = load_Bp(wv2p, VMID / 32, kb, tile, lane);
    const int half = lane >> 4, nn = tile * 16 + (lane & 15);
    const float bias = b2[nn];
#pragma unroll
    for (int mt = 0; mt < ENC_M / 16; ++mt) {
      v8f acc = {};
#pragma unroll
      for (int kb = 0; kb < VMID / 32; ++kb)
        acc = WMMA_BF16(acc, load_A(&sH1[mt * 16][0], VMID, kb * 32, lane), bf[kb]);
#pragma unroll
      for (int r = 0; r < 8; ++r) sH[mt * 16 + r + half * 8][nn] = acc[r] + bias;
    }
  }
  __syncthreads();

  // Fourier keys + segment sum: thread owns column j; flush atomics only on
  // (wave-uniform) batch change => ~n_segments atomics instead of per-row.
  {
    const int j = tid;
    const float tj8 = (float)j * (8.0f / (HID - 1));
    float aR = 0.f, aI = 0.f;
    int curb = batch[R0];
    int s0 = seg_start[curb];
    for (int r = 0; r < ENC_M; ++r) {
      const int R = R0 + r;
      const int b = batch[R];  // uniform across the block at each r
      if (b != curb) {
        atomicAdd(&z_re[curb * HID + j], aR);
        atomicAdd(&z_im[curb * HID + j], aI);
        aR = 0.f; aI = 0.f;
        curb = b;
        s0 = seg_start[b];
      }
      const float th = tj8 * (float)(R - s0);
      float s, c;
      __sincosf(th, &s, &c);
      const float h = sH[r][j];
      aR += h * c;
      aI += h * s;
    }
    atomicAdd(&z_re[curb * HID + j], aR);
    atomicAdd(&z_im[curb * HID + j], aI);
  }
}

// ---------------- size head + zc = z - (n_pred*card_w + card_b) ----------------
__global__ __launch_bounds__(SMID) void k_size(
    const float* __restrict__ w1, const float* __restrict__ b1,
    const float* __restrict__ ln_g, const float* __restrict__ ln_b,
    const float* __restrict__ w2, const float* __restrict__ b2,
    const float* __restrict__ card_w, const float* __restrict__ card_b,
    float* __restrict__ z_re, int* __restrict__ npred_i,
    float* __restrict__ out_mask, float* __restrict__ out_np) {
  __shared__ float sh[SMID];
  __shared__ float sMu, sVr, sNp;
  __shared__ int sNpi;
  const int b = blockIdx.x, j = threadIdx.x;
  const float* zr = z_re + (size_t)b * HID;
  float acc = b1[j];
  for (int k = 0; k < HID; ++k) acc += zr[k] * w1[k * SMID + j];
  sh[j] = acc;
  __syncthreads();
  if (j == 0) {
    float mu = 0.f;
    for (int i = 0; i < SMID; ++i) mu += sh[i];
    mu *= (1.f / SMID);
    float var = 0.f;
    for (int i = 0; i < SMID; ++i) { float d = sh[i] - mu; var += d * d; }
    sMu = mu; sVr = var * (1.f / SMID);
  }
  __syncthreads();
  const float xv = (sh[j] - sMu) * rsqrtf(sVr + 1e-5f) * ln_g[j] + ln_b[j];
  const float mv = mishf(xv) * w2[j];
  __syncthreads();
  sh[j] = mv;
  __syncthreads();
  if (j == 0) {
    float s = b2[0];
    for (int i = 0; i < SMID; ++i) s += sh[i];
    const float np = fmaxf(rintf(s), 0.f);  // round-half-even, clamp >= 0
    const int npi = (int)fminf(np, (float)NMAX);
    sNp = np; sNpi = npi;
    npred_i[b] = npi;
    out_np[b] = (float)npi;
  }
  __syncthreads();
  const float np = sNp;
  const int npi = sNpi;
  for (int idx = j; idx < HID; idx += SMID)
    z_re[(size_t)b * HID + idx] -= np * card_w[idx] + card_b[idx];
  if (j < NMAX) out_mask[b * NMAX + j] = (j < npi) ? 1.f : 0.f;
}

// ---------------- decoder: DEC_M (set,slot) rows (= 2 sets) per workgroup ----------------
__global__ __launch_bounds__(256) void k_decoder(
    const float* __restrict__ zc_re, const float* __restrict__ zc_im,
    const bf16_t* __restrict__ wd1p, const float* __restrict__ b1,
    const bf16_t* __restrict__ wd2p, const float* __restrict__ b2,
    const int* __restrict__ npred_i, float* __restrict__ out_xr) {
  __shared__ __align__(16) bf16_t sA[DEC_M][HID];    // 32KB
  __shared__ __align__(16) bf16_t sH1[DEC_M][DMID];  // 24KB
  __shared__ float sZr[2][HID], sZi[2][HID];         // 4KB
  const int tid = threadIdx.x, lane = tid & 31, wave = tid >> 5;
  const int R0 = blockIdx.x * DEC_M;
  const int b0 = R0 >> 5;  // first of the two sets

  for (int idx = tid; idx < 2 * HID; idx += 256) {
    const int s = idx >> 8, j = idx & (HID - 1);
    sZr[s][j] = zc_re[(size_t)(b0 + s) * HID + j];
    sZi[s][j] = zc_im[(size_t)(b0 + s) * HID + j];
  }
  __syncthreads();

  // zp.real = zc_re*cos - zc_im*sin; one sincos serves both sets (kk repeats)
  {
    const int j = tid;
    const float tj8 = (float)j * (8.f / (HID - 1));
    const float zr0 = sZr[0][j], zi0 = sZi[0][j];
    const float zr1 = sZr[1][j], zi1 = sZi[1][j];
    for (int kk = 0; kk < NMAX; ++kk) {
      float sn, cs;
      __sincosf(tj8 * (float)kk, &sn, &cs);
      sA[kk][j]        = (bf16_t)(zr0 * cs - zi0 * sn);
      sA[kk + NMAX][j] = (bf16_t)(zr1 * cs - zi1 * sn);
    }
  }
  __syncthreads();

  // GEMM1: [DEC_M x 256] @ [256 x 192]; B in regs across 4 M-tiles, mish fused
  for (int tile = wave; tile < DMID / 16; tile += 8) {
    v16bf bf[HID / 32];
#pragma unroll
    for (int kb = 0; kb < HID / 32; ++kb) bf[kb] = load_Bp(wd1p, HID / 32, kb, tile, lane);
    const int half = lane >> 4, nn = tile * 16 + (lane & 15);
    const float bias = b1[nn];
#pragma unroll
    for (int mt = 0; mt < DEC_M / 16; ++mt) {
      v8f acc = {};
#pragma unroll
      for (int kb = 0; kb < HID / 32; ++kb)
        acc = WMMA_BF16(acc, load_A(&sA[mt * 16][0], HID, kb * 32, lane), bf[kb]);
#pragma unroll
      for (int r = 0; r < 8; ++r)
        sH1[mt * 16 + r + half * 8][nn] = (bf16_t)mishf(acc[r] + bias);
    }
  }
  __syncthreads();

  // GEMM2: [DEC_M x 192] @ [192 x 128], masked store
  {
    v16bf bf[DMID / 32];
#pragma unroll
    for (int kb = 0; kb < DMID / 32; ++kb) bf[kb] = load_Bp(wd2p, DMID / 32, kb, wave, lane);
    const int half = lane >> 4, nn = wave * 16 + (lane & 15);
    const float bias = b2[nn];
    const int np0 = npred_i[b0], np1 = npred_i[b0 + 1];
#pragma unroll
    for (int mt = 0; mt < DEC_M / 16; ++mt) {
      v8f acc = {};
#pragma unroll
      for (int kb = 0; kb < DMID / 32; ++kb)
        acc = WMMA_BF16(acc, load_A(&sH1[mt * 16][0], DMID, kb * 32, lane), bf[kb]);
#pragma unroll
      for (int r = 0; r < 8; ++r) {
        const int m = mt * 16 + r + half * 8;
        const int kk = m & (NMAX - 1);
        const int np = (m >> 5) ? np1 : np0;
        const float v = acc[r] + bias;
        out_xr[(size_t)(R0 + m) * DIM + nn] = (kk < np) ? v : 0.f;
      }
    }
  }
}

extern "C" void kernel_launch(void* const* d_in, const int* in_sizes, int n_in,
                              void* d_out, int out_size, void* d_ws, size_t ws_size,
                              hipStream_t stream) {
  (void)in_sizes; (void)n_in; (void)out_size; (void)ws_size;
  const float* x        = (const float*)d_in[0];
  const int*   batch    = (const int*)d_in[1];  // jnp int64 w/o x64 => int32
  const float* rank_w   = (const float*)d_in[2];
  const float* rank_b   = (const float*)d_in[3];
  const float* val_w1   = (const float*)d_in[4];
  const float* val_b1   = (const float*)d_in[5];
  const float* val_ln_g = (const float*)d_in[6];
  const float* val_ln_b = (const float*)d_in[7];
  const float* val_w2   = (const float*)d_in[8];
  const float* val_b2   = (const float*)d_in[9];
  const float* card_w   = (const float*)d_in[10];
  const float* card_b   = (const float*)d_in[11];
  const float* size_w1  = (const float*)d_in[12];
  const float* size_b1  = (const float*)d_in[13];
  const float* size_lng = (const float*)d_in[14];
  const float* size_lnb = (const float*)d_in[15];
  const float* size_w2  = (const float*)d_in[16];
  const float* size_b2  = (const float*)d_in[17];
  const float* dec_w1   = (const float*)d_in[18];
  const float* dec_b1   = (const float*)d_in[19];
  const float* dec_w2   = (const float*)d_in[20];
  const float* dec_b2   = (const float*)d_in[21];

  // workspace carve-out (~4.7 MB)
  char* ws = (char*)d_ws;
  size_t off = 0;
  auto carve = [&](size_t bytes) -> char* {
    char* p = ws + off;
    off = (off + bytes + 255) & ~(size_t)255;
    return p;
  };
  float*  mag    = (float*)carve((size_t)NTOT * 4);
  int*    order  = (int*)carve((size_t)NTOT * 4);
  int*    segst  = (int*)carve((size_t)NBATCH * 4);
  int*    ncnt   = (int*)carve((size_t)NBATCH * 4);
  float*  z_re   = (float*)carve((size_t)NBATCH * HID * 4);
  float*  z_im   = (float*)carve((size_t)NBATCH * HID * 4);
  int*    npredi = (int*)carve((size_t)NBATCH * 4);
  bf16_t* wv1p   = (bf16_t*)carve((size_t)DIM * VMID * 2);
  bf16_t* wv2p   = (bf16_t*)carve((size_t)VMID * HID * 2);
  bf16_t* wd1p   = (bf16_t*)carve((size_t)HID * DMID * 2);
  bf16_t* wd2p   = (bf16_t*)carve((size_t)DMID * DIM * 2);

  float* out_xr   = (float*)d_out;                         // [NB,NMAX,DIM]
  float* out_mask = out_xr + (size_t)NBATCH * NMAX * DIM;  // [NB,NMAX]
  float* out_np   = out_mask + (size_t)NBATCH * NMAX;      // [NB]

  k_pack_w<<<(DIM * VMID + 255) / 256, 256, 0, stream>>>(val_w1, wv1p, DIM, VMID);
  k_pack_w<<<(VMID * HID + 255) / 256, 256, 0, stream>>>(val_w2, wv2p, VMID, HID);
  k_pack_w<<<(HID * DMID + 255) / 256, 256, 0, stream>>>(dec_w1, wd1p, HID, DMID);
  k_pack_w<<<(DMID * DIM + 255) / 256, 256, 0, stream>>>(dec_w2, wd2p, DMID, DIM);

  k_rank<<<NTOT / 8, 256, 0, stream>>>(x, rank_w, rank_b, mag);
  k_segbounds<<<(NBATCH + 255) / 256, 256, 0, stream>>>(batch, segst, ncnt);
  k_zinit<<<(NBATCH * HID + 255) / 256, 256, 0, stream>>>(ncnt, card_w, card_b, z_re, z_im);
  k_sort<<<NBATCH, 64, 0, stream>>>(mag, segst, ncnt, order);
  k_encoder<<<NTOT / ENC_M, 256, 0, stream>>>(x, batch, order, segst, wv1p, val_b1, val_ln_g,
                                              val_ln_b, wv2p, val_b2, z_re, z_im);
  k_size<<<NBATCH, SMID, 0, stream>>>(size_w1, size_b1, size_lng, size_lnb, size_w2, size_b2,
                                      card_w, card_b, z_re, npredi, out_mask, out_np);
  k_decoder<<<(NBATCH * NMAX) / DEC_M, 256, 0, stream>>>(z_re, z_im, wd1p, dec_b1, wd2p, dec_b2,
                                                         npredi, out_xr);
}
